// MoeGroupMLP_14663018348759
// MI455X (gfx1250) — matrix operand order repaired and checked by reading, not compiled, not run on hardware
//
#include <hip/hip_runtime.h>

#define TOKENS 2048
#define HIDDEN 1024
#define INTER  1024
#define NPAIRS (TOKENS * 2)
#define NEXP   8
#define LDS_STRIDE 1032   // 1024 halves + 8 pad -> row stride 2064B (16B aligned, bank-conflict free)

typedef _Float16 v16h __attribute__((ext_vector_type(16)));
typedef _Float16 v8h  __attribute__((ext_vector_type(8)));
typedef float    v8f  __attribute__((ext_vector_type(8)));
typedef int      v4i  __attribute__((ext_vector_type(4)));

typedef __attribute__((address_space(1))) v4i GV4;   // global int4
typedef __attribute__((address_space(3))) v4i LV4;   // LDS int4

static constexpr size_t WN = (size_t)NEXP * INTER * HIDDEN;   // per weight tensor (8M elems)
static constexpr size_t XN = (size_t)TOKENS * HIDDEN;         // activations (2M elems)

// -------- routing: deterministic offsets + stable scatter (no atomics) --------

__global__ void moe_offsets_kernel(const int* __restrict__ sel, int* __restrict__ offs) {
    int t = threadIdx.x;
    if (t > NEXP) return;
    int cnt = 0;
    for (int q = 0; q < NPAIRS; ++q) cnt += (sel[q] < t) ? 1 : 0;
    offs[t] = cnt;   // offs[e] = exclusive prefix; offs[8] = NPAIRS
}

__global__ void moe_scatter_kernel(const int* __restrict__ sel, int* __restrict__ pairlist) {
    int p = blockIdx.x * blockDim.x + threadIdx.x;
    if (p >= NPAIRS) return;
    int e = sel[p];
    int pos = 0;
    for (int q = 0; q < NPAIRS; ++q) {
        int s = sel[q];
        pos += ((s < e) || (s == e && q < p)) ? 1 : 0;
    }
    pairlist[pos] = p;
}

// -------- one-shot f32 -> f16 conversion of weights + activations into ws --------
// ws f16 layout: [gate | up | down | x], each source-order row-major.

__global__ __launch_bounds__(256)
void cvt_kernel(const float* __restrict__ g, const float* __restrict__ u,
                const float* __restrict__ d, const float* __restrict__ x,
                _Float16* __restrict__ out) {
    size_t i = ((size_t)blockIdx.x * blockDim.x + threadIdx.x) * 8;
    if (i >= 3 * WN + XN) return;
    const float* src;
    if      (i < WN)          src = g + i;
    else if (i < 2 * WN)      src = u + (i - WN);
    else if (i < 3 * WN)      src = d + (i - 2 * WN);
    else                      src = x + (i - 3 * WN);
    float4 f0 = ((const float4*)src)[0];
    float4 f1 = ((const float4*)src)[1];
    v8h h;
    h[0]=(_Float16)f0.x; h[1]=(_Float16)f0.y; h[2]=(_Float16)f0.z; h[3]=(_Float16)f0.w;
    h[4]=(_Float16)f1.x; h[5]=(_Float16)f1.y; h[6]=(_Float16)f1.z; h[7]=(_Float16)f1.w;
    *(v8h*)(out + i) = h;
}

// -------- fragment loaders --------

// A fragment (16x32 f16) from LDS tile. p = row*LDS_STRIDE + kk + (lane>=16 ? 16 : 0)
__device__ __forceinline__ v16h load_a16(const _Float16* p) {
    v8h lo = *(const v8h*)(p);
    v8h hi = *(const v8h*)(p + 8);
    return __builtin_shufflevector(lo, hi, 0,1,2,3,4,5,6,7,8,9,10,11,12,13,14,15);
}

// B fragment: 16 contiguous K halves (pre-converted f16 weights)
__device__ __forceinline__ v16h load_b_f16(const _Float16* p) {
    v8h lo = *(const v8h*)(p);
    v8h hi = *(const v8h*)(p + 8);
    return __builtin_shufflevector(lo, hi, 0,1,2,3,4,5,6,7,8,9,10,11,12,13,14,15);
}

// B fragment: 16 contiguous K values of one f32 weight row, converted on the fly (fallback)
__device__ __forceinline__ v16h load_b_f32(const float* p) {
    const float4* q = (const float4*)p;
    float4 f0 = q[0], f1 = q[1], f2 = q[2], f3 = q[3];
    v16h b;
    b[0]=(_Float16)f0.x; b[1]=(_Float16)f0.y; b[2]=(_Float16)f0.z; b[3]=(_Float16)f0.w;
    b[4]=(_Float16)f1.x; b[5]=(_Float16)f1.y; b[6]=(_Float16)f1.z; b[7]=(_Float16)f1.w;
    b[8]=(_Float16)f2.x; b[9]=(_Float16)f2.y; b[10]=(_Float16)f2.z; b[11]=(_Float16)f2.w;
    b[12]=(_Float16)f3.x; b[13]=(_Float16)f3.y; b[14]=(_Float16)f3.z; b[15]=(_Float16)f3.w;
    return b;
}

#define WMMA_F16(a, b, c) \
    __builtin_amdgcn_wmma_f32_16x16x32_f16(false, (a), false, (b), (short)0, (c), false, false)

// -------- fused expert MLP: 16 (token,k) rows per block, one expert --------

template <bool F16W>
__global__ __launch_bounds__(512)
void moe_mlp_kernel(const float* __restrict__ x,
                    const float* __restrict__ rw,
                    const float* __restrict__ gate_w,
                    const float* __restrict__ up_w,
                    const float* __restrict__ down_w,
                    const _Float16* __restrict__ wf16,   // f16: [gate|up|down|x]
                    const int*   __restrict__ offs,
                    const int*   __restrict__ pairlist,
                    float*       __restrict__ out) {
    __shared__ _Float16 x_lds[16 * LDS_STRIDE];   // 16 x 1024 activations (f16)
    __shared__ _Float16 h_lds[16 * LDS_STRIDE];   // 16 x 1024 SwiGLU intermediates (f16)
    __shared__ int   tokS[16];
    __shared__ float rwS[16];

    const int e      = blockIdx.y;
    const int segBeg = offs[e];
    const int segEnd = offs[e + 1];
    const int base   = segBeg + (int)blockIdx.x * 16;
    if (base >= segEnd) return;               // uniform early exit

    const int tid = threadIdx.x;
    if (tid < 16) {
        int idx = base + tid;
        if (idx < segEnd) {
            int p = pairlist[idx];
            tokS[tid] = p >> 1;               // token id
            rwS[tid]  = rw[p];                // router weight (flat [T, K])
        } else {
            tokS[tid] = -1;                   // padded row: never stored (rw=0)
            rwS[tid]  = 0.0f;
        }
    }
    __syncthreads();

    const int wave  = tid >> 5;
    const int lane  = tid & 31;
    const int lhalf = lane >> 4;              // 0 or 1
    const int l16   = lane & 15;
    const int akoff = lhalf * 16;             // A: K sub-block per lane half
    const int bkoff = lhalf * 16;             // B: K sub-block per lane half
    const int colbase = wave * 64;            // this wave's 64 columns

    // ---- stage x tile (16 x 1024 f16) into LDS ----
    if constexpr (F16W) {
        // Padded rows (tok<0) are mapped to token 0: WMMA rows are independent along M
        // and padded rows are never stored, so their contents are don't-care.
        const int row = wave;                               // one row per wave
        const int t   = (tokS[row] < 0) ? 0 : tokS[row];
        const _Float16* xrow = wf16 + 3 * WN + (size_t)t * HIDDEN;
#pragma unroll
        for (int it = 0; it < 4; ++it) {
            const int c = it * 256 + lane * 8;              // 8 halves = 16B per lane
#if __has_builtin(__builtin_amdgcn_global_load_async_to_lds_b128)
            __builtin_amdgcn_global_load_async_to_lds_b128(
                (GV4*)(uintptr_t)(xrow + c),
                (LV4*)(uint32_t)(uintptr_t)(&x_lds[row * LDS_STRIDE + c]),
                0, 0);
#else
            *(v8h*)(&x_lds[row * LDS_STRIDE + c]) = *(const v8h*)(xrow + c);
#endif
        }
#if __has_builtin(__builtin_amdgcn_global_load_async_to_lds_b128)
#if __has_builtin(__builtin_amdgcn_s_wait_asynccnt)
        __builtin_amdgcn_s_wait_asynccnt(0);
#else
        asm volatile("s_wait_asynccnt 0x0" ::: "memory");
#endif
#endif
    } else {
        // f32 source fallback: load + convert + ds-store
        const float4* x4 = (const float4*)x;
        for (int i = tid; i < 16 * (HIDDEN / 4); i += 512) {
            int row = i >> 8;
            int c4  = i & 255;
            int t   = tokS[row];
            float4 v = make_float4(0.f, 0.f, 0.f, 0.f);
            if (t >= 0) v = x4[(size_t)t * (HIDDEN / 4) + c4];
            _Float16* d = &x_lds[row * LDS_STRIDE + c4 * 4];
            d[0] = (_Float16)v.x; d[1] = (_Float16)v.y;
            d[2] = (_Float16)v.z; d[3] = (_Float16)v.w;
        }
    }
    __syncthreads();

    // ---- phase 1: h = silu(x Wg^T) * (x Wu^T); 4 N-tiles live so each A feeds 8 WMMAs ----
    {
        v8f accg[4], accu[4];
        const float*    gp32[4]; const float*    up32[4];
        const _Float16* gp16[4]; const _Float16* up16[4];
#pragma unroll
        for (int nt = 0; nt < 4; ++nt) {
            const int col = colbase + nt * 16 + l16;
            const size_t roff = ((size_t)e * INTER + col) * HIDDEN + bkoff;
            if constexpr (F16W) {
                gp16[nt] = wf16 + roff;
                up16[nt] = wf16 + WN + roff;
            } else {
                gp32[nt] = gate_w + roff;
                up32[nt] = up_w + roff;
            }
            accg[nt] = (v8f){};
            accu[nt] = (v8f){};
        }
        for (int kk = 0; kk < HIDDEN; kk += 32) {
            v16h a = load_a16(&x_lds[l16 * LDS_STRIDE + kk + akoff]);
#pragma unroll
            for (int nt = 0; nt < 4; ++nt) {
                v16h bg, bu;
                if constexpr (F16W) { bg = load_b_f16(gp16[nt] + kk); bu = load_b_f16(up16[nt] + kk); }
                else                { bg = load_b_f32(gp32[nt] + kk); bu = load_b_f32(up32[nt] + kk); }
                accg[nt] = WMMA_F16(a, bg, accg[nt]);
                accu[nt] = WMMA_F16(a, bu, accu[nt]);
            }
        }
#pragma unroll
        for (int nt = 0; nt < 4; ++nt) {
            const int col = colbase + nt * 16 + l16;
#pragma unroll
            for (int r = 0; r < 8; ++r) {
                float g = accg[nt][r];
                float u = accu[nt][r];
                // silu via v_exp + v_rcp (no IEEE divide expansion)
                float s = g * __builtin_amdgcn_rcpf(1.0f + __expf(-g));
                int hrow = lhalf * 8 + r;                 // C layout: M = r + 8*(lane>=16)
                h_lds[hrow * LDS_STRIDE + col] = (_Float16)(s * u);
            }
        }
    }
    __syncthreads();

    // ---- phase 2: out_tile = h @ Wd^T over columns [colbase, colbase+64) ----
    v8f acc[4];
    const float*    dp32[4];
    const _Float16* dp16[4];
#pragma unroll
    for (int nt = 0; nt < 4; ++nt) {
        const size_t roff = ((size_t)e * HIDDEN + colbase + nt * 16 + l16) * INTER + bkoff;
        if constexpr (F16W) dp16[nt] = wf16 + 2 * WN + roff;
        else                dp32[nt] = down_w + roff;
        acc[nt] = (v8f){};
    }
    for (int kk = 0; kk < INTER; kk += 32) {
        v16h a = load_a16(&h_lds[l16 * LDS_STRIDE + kk + akoff]);
#pragma unroll
        for (int nt = 0; nt < 4; ++nt) {
            v16h b;
            if constexpr (F16W) b = load_b_f16(dp16[nt] + kk);
            else                b = load_b_f32(dp32[nt] + kk);
            acc[nt] = WMMA_F16(a, b, acc[nt]);
        }
    }

    // scatter-add with router weight; TOPK=2 -> <=2 fp adds per element from 0 => order-independent
#pragma unroll
    for (int r = 0; r < 8; ++r) {
        int m = lhalf * 8 + r;
        int t = tokS[m];
        if (t >= 0) {
            float w = rwS[m];
            float* o = out + (size_t)t * HIDDEN;
#pragma unroll
            for (int nt = 0; nt < 4; ++nt) {
                atomicAdd(o + colbase + nt * 16 + l16, w * acc[nt][r]);
            }
        }
    }
}

// -------- launcher --------

extern "C" void kernel_launch(void* const* d_in, const int* in_sizes, int n_in,
                              void* d_out, int out_size, void* d_ws, size_t ws_size,
                              hipStream_t stream) {
    const float* x    = (const float*)d_in[0];   // [2048, 1024]
    const float* rw   = (const float*)d_in[1];   // [2048, 2]
    const int*   sel  = (const int*)  d_in[2];   // [2048, 2]
    // d_in[3] = token_per_expert : unused by the reference math
    const float* gw   = (const float*)d_in[4];   // [8, 1024, 1024]
    const float* uw   = (const float*)d_in[5];   // [8, 1024, 1024]
    const float* dw   = (const float*)d_in[6];   // [8, 1024, 1024]
    float* out = (float*)d_out;                  // [2048, 1024]

    int* wsI      = (int*)d_ws;
    int* offs     = wsI;        // 9 ints
    int* pairlist = wsI + 16;   // 4096 ints
    _Float16* f16buf = (_Float16*)((char*)d_ws + 32768);  // [gate|up|down|x] f16, ~52 MB

    const size_t need_f16 = 32768 + (3 * WN + XN) * sizeof(_Float16);
    const bool use_f16 = (ws_size >= need_f16);           // fixed per session -> deterministic

    (void)hipMemsetAsync(out, 0, (size_t)out_size * sizeof(float), stream);
    moe_offsets_kernel<<<1, 16, 0, stream>>>(sel, offs);
    moe_scatter_kernel<<<NPAIRS / 256, 256, 0, stream>>>(sel, pairlist);

    if (use_f16) {
        const int groups = (int)((3 * WN + XN) / 8);      // 3,407,872 (multiple of 256)
        cvt_kernel<<<groups / 256, 256, 0, stream>>>(gw, uw, dw, x, f16buf);
        moe_mlp_kernel<true><<<dim3(256, 8), 512, 0, stream>>>(
            x, rw, gw, uw, dw, f16buf, offs, pairlist, out);
    } else {
        moe_mlp_kernel<false><<<dim3(256, 8), 512, 0, stream>>>(
            x, rw, gw, uw, dw, nullptr, offs, pairlist, out);
    }
    (void)in_sizes; (void)n_in; (void)ws_size;
}
